// ColumnSelfAttention_12472585027591
// MI455X (gfx1250) — matrix operand order repaired and checked by reading, not compiled, not run on hardware
//
#include <hip/hip_runtime.h>
#include <hip/hip_bf16.h>

// ---------------------------------------------------------------------------
// ColumnSelfAttention for MI455X (gfx1250): bf16 WMMA pipeline, fp32 accum.
//   R=128 rows, C=256 cols, B=1, E=768, H=12 heads, D=64 head dim.
//   q,k,v = GEMM(x, w*) -> bf16 scratch; per (h,c): S=QK^T, masked softmax,
//   probs out (fp32), ctx = P V -> bf16 scratch; out = GEMM(ctx, wo)+bo.
// Uses v_wmma_f32_16x16x32_bf16 + gfx1250 async global->LDS DMA for the
// conversion-free staging paths.
// ---------------------------------------------------------------------------

typedef __attribute__((ext_vector_type(16))) __bf16 v16bf;
typedef __attribute__((ext_vector_type(8)))  float  v8f;
typedef int v4i __attribute__((vector_size(16)));
typedef __attribute__((address_space(1))) v4i* GV4P;  // global int4*
typedef __attribute__((address_space(3))) v4i* LV4P;  // LDS int4*

#define R_DIM 128
#define C_DIM 256
#define H_DIM 12
#define D_DIM 64
#define E_DIM 768
#define M_DIM (R_DIM * C_DIM)   // 32768 (B==1)
#define SCALING 0.125f          // 64^-0.5

__device__ __forceinline__ unsigned short f2bf(float f) {
  unsigned u = __float_as_uint(f);
  u += 0x7FFFu + ((u >> 16) & 1u);          // round-to-nearest-even
  return (unsigned short)(u >> 16);
}
__device__ __forceinline__ unsigned pack2bf(float lo, float hi) {
  return (unsigned)f2bf(lo) | ((unsigned)f2bf(hi) << 16);
}
__device__ __forceinline__ v8f vzero8() {
  v8f z;
#pragma unroll
  for (int i = 0; i < 8; ++i) z[i] = 0.0f;
  return z;
}

// WMMA wrapper: operands held as 8 u32 (= 16 packed bf16) per lane, matching
// the ISA VGPR layout for 16-bit A (16x32) / B (32x16) matrices.
__device__ __forceinline__ v8f wmma_bf16(const unsigned (&au)[8],
                                         const unsigned (&bu)[8], v8f c) {
  union U { unsigned u[8]; v16bf v; } A, B;
#pragma unroll
  for (int i = 0; i < 8; ++i) { A.u[i] = au[i]; B.u[i] = bu[i]; }
  return __builtin_amdgcn_wmma_f32_16x16x32_bf16(
      /*neg_a=*/false, A.v, /*neg_b=*/false, B.v,
      /*c_mod=*/(short)0, c, /*reuse_a=*/false, /*reuse_b=*/false);
}

// ---------------------------------------------------------------------------
// GEMM: Out[M,N] = (A[M,K] @ W[N,K]^T + bias[N]) * scale
// A is fp32 (converted to bf16 in LDS) or bf16 (A_BF16 -> async DMA to LDS).
// Block = 256 threads (8 waves). Tile 128x64, K-step 32. 4 WMMA / wave / step.
// ---------------------------------------------------------------------------
template <bool A_BF16, bool OUT_BF16>
__global__ __launch_bounds__(256)
void gemm_bias_kernel(const void* __restrict__ Ap, const float* __restrict__ W,
                      const float* __restrict__ bias, void* __restrict__ Outp,
                      int M, int N, int K, float scale) {
  __shared__ unsigned As[128 * 20];   // 128 x (16 k-pairs + 4 pad) u32
  __shared__ unsigned Bs[64 * 20];

  const int t = threadIdx.x;
  const int lane = t & 31, wid = t >> 5;
  const int wm = wid & 3, wn = wid >> 2;       // 4x2 wave grid -> 32x32 each
  const int lm = lane & 15, half = lane >> 4;
  const int m0 = blockIdx.x * 128, n0 = blockIdx.y * 64;

  v8f acc[2][2];
#pragma unroll
  for (int i = 0; i < 2; ++i)
#pragma unroll
    for (int j = 0; j < 2; ++j) acc[i][j] = vzero8();

  const float*    Af = (const float*)Ap;
  const unsigned* Ab = (const unsigned*)Ap;

  for (int k0 = 0; k0 < K; k0 += 32) {
    // ---- stage A tile 128x32 (bf16 pairs) ----
    if constexpr (A_BF16) {
      // pure copy: async DMA 16B/transfer, both sides 16B aligned
#pragma unroll
      for (int i = 0; i < 2; ++i) {
        int u = t + i * 256;                 // 0..511 = 128 rows x 4 x b128
        int row = u >> 2, cq = (u & 3) * 4;  // cq: u32 index, 16B granules
        __builtin_amdgcn_global_load_async_to_lds_b128(
            (GV4P)(Ab + (size_t)(m0 + row) * (K >> 1) + (k0 >> 1) + cq),
            (LV4P)(As + row * 20 + cq), 0, 0);
      }
    } else {
#pragma unroll
      for (int i = 0; i < 8; ++i) {
        int u = t + i * 256, row = u >> 4, kp = u & 15;
        float2 f = ((const float2*)Af)[((size_t)(m0 + row) * K + k0) / 2 + kp];
        As[row * 20 + kp] = pack2bf(f.x, f.y);
      }
    }
    // ---- stage W tile 64x32 (fp32 -> bf16 pack) ----
#pragma unroll
    for (int i = 0; i < 4; ++i) {
      int u = t + i * 256, row = u >> 4, kp = u & 15;
      float2 f = ((const float2*)W)[((size_t)(n0 + row) * K + k0) / 2 + kp];
      Bs[row * 20 + kp] = pack2bf(f.x, f.y);
    }
    if (k0 + 32 < K) {  // hint next A tile into cache (global_prefetch_b8)
      __builtin_prefetch((const char*)Ap +
                         ((size_t)(m0 + (t >> 1)) * K + k0 + 32) *
                             (A_BF16 ? 2 : 4), 0, 1);
    }
    if constexpr (A_BF16) __builtin_amdgcn_s_wait_asynccnt(0);
    __syncthreads();

    unsigned af[2][8], bfr[2][8];
#pragma unroll
    for (int sm = 0; sm < 2; ++sm) {
      int arow = wm * 32 + sm * 16 + lm;
#pragma unroll
      for (int v = 0; v < 8; ++v) {  // A layout: k/2 = (v<4?v:8+(v-4)) + 4*half
        int kp = (v < 4 ? v : 8 + (v - 4)) + 4 * half;
        af[sm][v] = As[arow * 20 + kp];
      }
    }
#pragma unroll
    for (int sn = 0; sn < 2; ++sn) {
      int brow = wn * 32 + sn * 16 + lm;
#pragma unroll
      for (int v = 0; v < 8; ++v) {  // B layout: k/2 = (v<4?v:8+(v-4)) + 4*half
        bfr[sn][v] = Bs[brow * 20 + (v < 4 ? v : 8 + (v - 4)) + 4 * half];
      }
    }
#pragma unroll
    for (int sm = 0; sm < 2; ++sm)
#pragma unroll
      for (int sn = 0; sn < 2; ++sn)
        acc[sm][sn] = wmma_bf16(af[sm], bfr[sn], acc[sm][sn]);
    __syncthreads();
  }

  // epilogue: (acc + bias) * scale ; D layout: m = v + 8*half, n = lm
#pragma unroll
  for (int sm = 0; sm < 2; ++sm)
#pragma unroll
    for (int sn = 0; sn < 2; ++sn) {
      int gn = n0 + wn * 32 + sn * 16 + lm;
      float b = bias[gn];
#pragma unroll
      for (int v = 0; v < 8; ++v) {
        int gm = m0 + wm * 32 + sm * 16 + v + 8 * half;
        float val = (acc[sm][sn][v] + b) * scale;
        if constexpr (OUT_BF16)
          ((unsigned short*)Outp)[(size_t)gm * N + gn] = f2bf(val);
        else
          ((float*)Outp)[(size_t)gm * N + gn] = val;
      }
    }
}

// ---------------------------------------------------------------------------
// Attention over rows, one block per (c, h). 8 waves; wave w owns S-strip
// rows [16w, 16w+16). Softmax per row via 8 regs x 16-lane shfl_xor tree.
// Q/K staged via async global->LDS DMA (pure bf16 copies); V staged manually
// (needs a transpose for the B-operand layout of the P*V WMMA).
// ---------------------------------------------------------------------------
__global__ __launch_bounds__(256)
void col_attn_kernel(const unsigned short* __restrict__ qb,
                     const unsigned short* __restrict__ kb,
                     const unsigned short* __restrict__ vb,
                     const unsigned char* __restrict__ mask,  // [B=1, C] bool
                     float* __restrict__ probs,               // [H,C,R,R] f32
                     unsigned short* __restrict__ ctx) {      // [M,E] bf16
  __shared__ unsigned       sQK[2 * 128 * 36];  // Q,K staged; reused for P
  __shared__ unsigned short sVt[64 * 136];      // V^T: [d][j], pad 8
  unsigned* Qs = sQK;
  unsigned* Ks = sQK + 128 * 36;
  unsigned* Ps = sQK;                           // 128*68 u32 <= 2*128*36

  const int c = blockIdx.x, h = blockIdx.y;
  const int t = threadIdx.x;
  const int lane = t & 31, wid = t >> 5;
  const int lm = lane & 15, half = lane >> 4;
  const int istrip = wid * 16;

  const unsigned* q32 = (const unsigned*)qb;
  const unsigned* k32 = (const unsigned*)kb;
  const unsigned* v32 = (const unsigned*)vb;

  // async-stage Q,K: 128 rows x 32 u32, b128 granules (16B aligned both sides)
#pragma unroll
  for (int i = 0; i < 4; ++i) {
    int u = t + i * 256;                 // 0..1023
    int r = u >> 3, cq = (u & 7) * 4;    // u32 col within head tile
    size_t g = (size_t)(r * C_DIM + c) * (E_DIM >> 1) + h * 32 + cq;
    __builtin_amdgcn_global_load_async_to_lds_b128(
        (GV4P)(q32 + g), (LV4P)(Qs + r * 36 + cq), 0, 0);
    __builtin_amdgcn_global_load_async_to_lds_b128(
        (GV4P)(k32 + g), (LV4P)(Ks + r * 36 + cq), 0, 0);
  }
  // stage V transposed: [d][j]
#pragma unroll
  for (int i = 0; i < 16; ++i) {
    int u = t + i * 256;                 // 0..4095 = 128 rows x 32 u32
    int r = u >> 5, cp = u & 31;
    unsigned vv = v32[(size_t)(r * C_DIM + c) * (E_DIM >> 1) + h * 32 + cp];
    sVt[(2 * cp) * 136 + r]     = (unsigned short)(vv & 0xFFFFu);
    sVt[(2 * cp + 1) * 136 + r] = (unsigned short)(vv >> 16);
  }
  __builtin_amdgcn_s_wait_asynccnt(0);
  __syncthreads();

  // S = Q K^T  (K-dim = 64 -> two k-steps of 32)
  v8f acc[8];
#pragma unroll
  for (int jt = 0; jt < 8; ++jt) acc[jt] = vzero8();
#pragma unroll
  for (int s = 0; s < 2; ++s) {
    unsigned af[8];
#pragma unroll
    for (int v = 0; v < 8; ++v) {
      int kp = (v < 4 ? v : 8 + (v - 4)) + 4 * half + 16 * s;
      af[v] = Qs[(istrip + lm) * 36 + kp];
    }
#pragma unroll
    for (int jt = 0; jt < 8; ++jt) {
      unsigned bfr[8];
#pragma unroll
      for (int v = 0; v < 8; ++v)
        bfr[v] = Ks[(jt * 16 + lm) * 36 + (v < 4 ? v : 8 + (v - 4)) + 4 * half + 16 * s];
      acc[jt] = wmma_bf16(af, bfr, acc[jt]);
    }
  }

  if (mask[c] != 0) {  // padded column: all logits -> -10000 (uniform softmax)
#pragma unroll
    for (int jt = 0; jt < 8; ++jt)
#pragma unroll
      for (int v = 0; v < 8; ++v) acc[jt][v] = -10000.0f;
  }

  // row softmax: row i = istrip + v + 8*half spans 8 tiles x 16 lanes
#pragma unroll
  for (int v = 0; v < 8; ++v) {
    float mx = acc[0][v];
#pragma unroll
    for (int jt = 1; jt < 8; ++jt) mx = fmaxf(mx, acc[jt][v]);
    mx = fmaxf(mx, __shfl_xor(mx, 1));
    mx = fmaxf(mx, __shfl_xor(mx, 2));
    mx = fmaxf(mx, __shfl_xor(mx, 4));
    mx = fmaxf(mx, __shfl_xor(mx, 8));
    float sum = 0.0f;
#pragma unroll
    for (int jt = 0; jt < 8; ++jt) {
      float e = __expf(acc[jt][v] - mx);
      acc[jt][v] = e;
      sum += e;
    }
    sum += __shfl_xor(sum, 1);
    sum += __shfl_xor(sum, 2);
    sum += __shfl_xor(sum, 4);
    sum += __shfl_xor(sum, 8);
    float inv = 1.0f / sum;
#pragma unroll
    for (int jt = 0; jt < 8; ++jt) acc[jt][v] *= inv;
  }

  // attn_probs out (fp32): [h][c][i][j]
  {
    size_t base = (size_t)(h * C_DIM + c) * R_DIM * R_DIM;
#pragma unroll
    for (int jt = 0; jt < 8; ++jt)
#pragma unroll
      for (int v = 0; v < 8; ++v)
        probs[base + (size_t)(istrip + v + 8 * half) * R_DIM + jt * 16 + lm] =
            acc[jt][v];
  }

  __syncthreads();  // everyone done reading Qs/Ks -> recycle as P
  unsigned short* Ph = (unsigned short*)Ps;  // stride 136 bf16 (=68 u32)
#pragma unroll
  for (int jt = 0; jt < 8; ++jt)
#pragma unroll
    for (int v = 0; v < 8; ++v)
      Ph[(istrip + v + 8 * half) * 136 + jt * 16 + lm] = f2bf(acc[jt][v]);
  __syncthreads();

  // context strip = P[16x128] @ V[128x64]  (K-dim = 128 -> four k-steps)
  v8f acc2[4];
#pragma unroll
  for (int dt = 0; dt < 4; ++dt) acc2[dt] = vzero8();
#pragma unroll
  for (int s = 0; s < 4; ++s) {
    unsigned af[8];
#pragma unroll
    for (int v = 0; v < 8; ++v) {
      int kp = (v < 4 ? v : 8 + (v - 4)) + 4 * half + 16 * s;  // j/2
      af[v] = Ps[(istrip + lm) * 68 + kp];
    }
#pragma unroll
    for (int dt = 0; dt < 4; ++dt) {
      unsigned bfr[8];
#pragma unroll
      for (int v = 0; v < 8; ++v) {
        int kp = (v < 4 ? v : 8 + (v - 4)) + 4 * half + 16 * s;  // j/2
        bfr[v] = *(const unsigned*)&sVt[(dt * 16 + lm) * 136 + 2 * kp];
      }
      acc2[dt] = wmma_bf16(af, bfr, acc2[dt]);
    }
  }

  // ctx (bf16) to scratch [M][E]: row = i*C + c, col = h*64 + d
#pragma unroll
  for (int dt = 0; dt < 4; ++dt)
#pragma unroll
    for (int v = 0; v < 8; ++v) {
      int i = istrip + v + 8 * half;
      int d = dt * 16 + lm;
      ctx[(size_t)(i * C_DIM + c) * E_DIM + h * 64 + d] = f2bf(acc2[dt][v]);
    }
}

// ---------------------------------------------------------------------------
extern "C" void kernel_launch(void* const* d_in, const int* in_sizes, int n_in,
                              void* d_out, int out_size, void* d_ws,
                              size_t ws_size, hipStream_t stream) {
  const float*         x   = (const float*)d_in[0];
  const unsigned char* pad = (const unsigned char*)d_in[1];  // jax bool = u8
  const float* wq = (const float*)d_in[2];
  const float* bq = (const float*)d_in[3];
  const float* wk = (const float*)d_in[4];
  const float* bk = (const float*)d_in[5];
  const float* wv = (const float*)d_in[6];
  const float* bv = (const float*)d_in[7];
  const float* wo = (const float*)d_in[8];
  const float* bo = (const float*)d_in[9];
  (void)in_sizes; (void)n_in; (void)out_size; (void)ws_size;

  const int M = M_DIM, N = E_DIM, K = E_DIM;
  const size_t elems = (size_t)M * E_DIM;            // 25,165,824
  unsigned short* qbuf = (unsigned short*)d_ws;      // 4 x 48 MiB bf16 scratch
  unsigned short* kbuf = qbuf + elems;
  unsigned short* vbuf = kbuf + elems;
  unsigned short* cbuf = vbuf + elems;

  float* out   = (float*)d_out;                      // [R,C,B,E] fp32
  float* probs = out + elems;                        // [H,C,B,R,R] fp32

  dim3 blk(256);
  dim3 ggrid(M / 128, N / 64);                       // 256 x 12
  gemm_bias_kernel<false, true><<<ggrid, blk, 0, stream>>>(
      x, wq, bq, qbuf, M, N, K, SCALING);            // q = (xWq^T+bq)*scale
  gemm_bias_kernel<false, true><<<ggrid, blk, 0, stream>>>(
      x, wk, bk, kbuf, M, N, K, 1.0f);
  gemm_bias_kernel<false, true><<<ggrid, blk, 0, stream>>>(
      x, wv, bv, vbuf, M, N, K, 1.0f);

  dim3 agrid(C_DIM, H_DIM);                          // 256 x 12 blocks
  col_attn_kernel<<<agrid, blk, 0, stream>>>(qbuf, kbuf, vbuf, pad, probs,
                                             cbuf);

  gemm_bias_kernel<true, false><<<ggrid, blk, 0, stream>>>(
      cbuf, wo, bo, out, M, N, K, 1.0f);             // out = ctx Wo^T + bo
}
